// GRU_44848048505578
// MI455X (gfx1250) — compile-verified, matrix operand order checked by reading, not run on hardware
//
#include <hip/hip_runtime.h>
#include <hip/hip_bf16.h>
#include <math.h>

// GRU: B=32, T=2048, D=H=256, L=2.  Critical path = 2*2048 sequential steps.
// 4 workgroups (one per WGP) each own a 64-wide hidden slice; weight slices
// live in LDS pre-swizzled into the exact WMMA B-fragment layout (one
// contiguous v16bf per lane -> 2x ds_load_b128).  Per step only the
// h-projection is on the critical chain; x(t+1) is DMA'd into LDS by the
// Tensor Data Mover and projected inside the cross-WG barrier shadow.
// Projections are 1x3 register-blocked (A fragment reused across 3 N-tiles).

typedef __attribute__((ext_vector_type(16))) __bf16 v16bf;
typedef __attribute__((ext_vector_type(8)))  __bf16 v8bf;
typedef __attribute__((ext_vector_type(4)))  __bf16 v4bf;
typedef __attribute__((ext_vector_type(8)))  float  v8f;
typedef __attribute__((ext_vector_type(4)))  unsigned int v4u;
typedef __attribute__((ext_vector_type(8)))  int v8i;
typedef __attribute__((ext_vector_type(4)))  int v4i;

#define B_ 32
#define T_ 2048
#define D_ 256
#define H_ 256
#define L_ 2
#define G_ (3 * H_)      // 768 gate columns
#define NWG 4            // workgroups cooperating on the scan
#define SLICE (H_ / NWG) // 64 hidden units per WG
#define NC (3 * SLICE)   // 192 local gate columns per WG
#define NT (NC / 16)     // 12 N-tiles per WG
#define KK (D_ / 32)     // 8 k-steps per tile
#define THREADS 256      // 8 waves (wave32)
// Byte offset of sX inside the (dynamic, base-0) LDS allocation.
#define SX_LDS_OFF (2u * NT * KK * 32 * 16 * 2)

// ---------------------------------------------------------------- weights ---
__global__ void gru_wcvt(const float* __restrict__ Wih, const float* __restrict__ Whh,
                         __bf16* __restrict__ WihT, __bf16* __restrict__ WhhT) {
  int idx = blockIdx.x * blockDim.x + threadIdx.x;
  const int total = L_ * G_ * D_;
  if (idx >= total) return;
  int l   = idx / (G_ * D_);
  int rem = idx % (G_ * D_);
  int g   = rem / D_;
  int k   = rem % D_;
  WihT[(size_t)l * D_ * G_ + (size_t)k * G_ + g] = (__bf16)Wih[idx];
  WhhT[(size_t)l * H_ * G_ + (size_t)k * G_ + g] = (__bf16)Whh[idx];
}

// -------------------------------------------------------------- x -> bf16 ---
__global__ void gru_xcvt(const float* __restrict__ xin, __bf16* __restrict__ xout,
                         int n4) {
  int i = blockIdx.x * blockDim.x + threadIdx.x;
  if (i >= n4) return;
  float4 f = ((const float4*)xin)[i];
  v4bf o;
  o[0] = (__bf16)f.x; o[1] = (__bf16)f.y; o[2] = (__bf16)f.z; o[3] = (__bf16)f.w;
  ((v4bf*)xout)[i] = o;
}

// ------------------------------------------------------------------- init ---
__global__ void gru_init(float* __restrict__ hbuf, unsigned* __restrict__ cnt) {
  int idx = blockIdx.x * blockDim.x + threadIdx.x;
  if (idx < 2 * B_ * H_) hbuf[idx] = 1.0f;  // h0 = ones (both parities)
  if (idx == 0) *cnt = 0u;
}

// --------------------------------------------- TDM: 32x256 bf16 tile load ---
// D# per CDNA5 ISA ch.8: group0 = {flags, lds_addr, global_addr, type=2},
// group1 = {data_size=2B, tensor_dim0=256, tensor_dim1=32, tile 256x32,
// dim0_stride=T*D}; groups 2/3 zero (tensor <= 2D).  Tracked by TENSORcnt.
__device__ __forceinline__ void tdm_load_tile(const __bf16* gsrc, unsigned lds_off) {
  unsigned long long ga = (unsigned long long)(uintptr_t)gsrc;
  unsigned lo = (unsigned)__builtin_amdgcn_readfirstlane((int)(unsigned)ga);
  unsigned hi = (unsigned)__builtin_amdgcn_readfirstlane((int)(unsigned)(ga >> 32));
  v4u g0;
  g0[0] = 1u;                                   // count=1 valid user descriptor
  g0[1] = lds_off;                              // D#.lds_addr (bytes)
  g0[2] = lo;                                   // D#.global_addr[31:0]
  g0[3] = (hi & 0x01FFFFFFu) | (2u << 30);      // global_addr[56:32] | type=2
  v8i g1;
  g1[0] = 1 << 16;                              // wg_mask=0, data_size=1 (2B)
  g1[1] = (D_ & 0xFFFF) << 16;                  // tensor_dim0[15:0]
  g1[2] = (D_ >> 16) | ((B_ & 0xFFFF) << 16);   // tensor_dim0[31:16]|td1[15:0]
  g1[3] = (B_ >> 16) | (D_ << 16);              // td1[31:16] | tile_dim0=256
  g1[4] = B_;                                   // tile_dim1=32, tile_dim2=0
  g1[5] = T_ * D_;                              // tensor_dim0_stride[31:0]
  g1[6] = 0;                                    // stride0[47:32], stride1 lo
  g1[7] = 0;
  v4i gz4 = {0, 0, 0, 0};
  v8i gz8 = {0, 0, 0, 0, 0, 0, 0, 0};
  // 6-arg form (clang-23 / therock-10.0 headers): trailing group + cpol.
  __builtin_amdgcn_tensor_load_to_lds(g0, g1, gz4, gz4, gz8, 0);
}

// ------------------------------------------------- 1x3 blocked projection ---
__device__ __forceinline__ void proj3(const __bf16* __restrict__ Am,
                                      const v16bf* __restrict__ Bf,
                                      float* __restrict__ Cm,
                                      int wave, int lane) {
  const int m0    = (wave & 1) * 16;
  const int ntb   = (wave >> 1) * 3;
  const int row   = m0 + (lane & 15);
  const int khalf = (lane >> 4) << 3;
  v8f acc0 = {}, acc1 = {}, acc2 = {};
#pragma unroll
  for (int kk = 0; kk < KK; ++kk) {
    const __bf16* ap = Am + row * D_ + kk * 32 + khalf;
    v8bf a0 = *(const v8bf*)ap;          // K pairs 0..7   (ds_load_b128)
    v8bf a1 = *(const v8bf*)(ap + 16);   // K pairs 16..23 (ds_load_b128)
    v16bf af = __builtin_shufflevector(a0, a1, 0, 1, 2, 3, 4, 5, 6, 7,
                                       8, 9, 10, 11, 12, 13, 14, 15);
    v16bf b0 = Bf[((ntb + 0) * KK + kk) * 32 + lane];
    v16bf b1 = Bf[((ntb + 1) * KK + kk) * 32 + lane];
    v16bf b2 = Bf[((ntb + 2) * KK + kk) * 32 + lane];
    acc0 = __builtin_amdgcn_wmma_f32_16x16x32_bf16(false, af, false, b0,
                                                   (short)0, acc0, false, false);
    acc1 = __builtin_amdgcn_wmma_f32_16x16x32_bf16(false, af, false, b1,
                                                   (short)0, acc1, false, false);
    acc2 = __builtin_amdgcn_wmma_f32_16x16x32_bf16(false, af, false, b2,
                                                   (short)0, acc2, false, false);
  }
  const int coln = lane & 15;
  const int mrow = m0 + ((lane >> 4) << 3);  // C: vgpr r -> M = m0+r(+8)
#pragma unroll
  for (int r = 0; r < 8; ++r) {
    Cm[(mrow + r) * NC + (ntb + 0) * 16 + coln] = acc0[r];
    Cm[(mrow + r) * NC + (ntb + 1) * 16 + coln] = acc1[r];
    Cm[(mrow + r) * NC + (ntb + 2) * 16 + coln] = acc2[r];
  }
}

// ------------------------------------------------------------------- scan ---
__global__ void __launch_bounds__(THREADS, 1)
gru_scan(const __bf16* __restrict__ x, const __bf16* __restrict__ WihT,
         const __bf16* __restrict__ WhhT, const float* __restrict__ bih,
         const float* __restrict__ bhh, float* __restrict__ hbuf,
         unsigned* __restrict__ cnt, float* __restrict__ yf,
         __bf16* __restrict__ ybf) {
  extern __shared__ char smem[];
  __bf16* sWihF = (__bf16*)smem;             // [NT][KK][32][16] bf16 (96 KB)
  __bf16* sWhhF = sWihF + NT * KK * 32 * 16; // [NT][KK][32][16] bf16 (96 KB)
  __bf16* sX    = sWhhF + NT * KK * 32 * 16; // [32][256] bf16 (16 KB), TDM dst
  __bf16* sH    = sX + B_ * D_;              // [32][256] bf16 (16 KB)
  float*  sXP   = (float*)(sH + B_ * H_);    // [32][192] f32  (24 KB)
  float*  sHP   = sXP + B_ * NC;             // [32][192] f32  (24 KB)
  float*  sBih  = sHP + B_ * NC;             // [192] f32
  float*  sBhh  = sBih + NC;                 // [192] f32

  const int tid  = threadIdx.x;
  const int wg   = blockIdx.x;   // 0..3 -> hidden slice
  const int lane = tid & 31;
  const int wave = tid >> 5;     // 0..7

  // One-time: swizzle this WG's weight-column slices into fragment layout.
  for (int i = tid; i < NT * KK * 32 * 16; i += THREADS) {
    int e   = i & 15;
    int ln  = (i >> 4) & 31;
    int kk  = (i >> 9) & (KK - 1);
    int nt  = i >> 12;
    int v   = e >> 1, odd = e & 1;
    int k0  = kk * 32 + ((ln >> 4) << 3) + ((v < 4) ? 2 * v : 16 + 2 * (v - 4)) + odd;
    int c   = nt * 16 + (ln & 15);
    int g   = (c / SLICE) * H_ + wg * SLICE + (c % SLICE);  // {r,z,n} gate col
    sWihF[i] = WihT[(size_t)k0 * G_ + g];
    sWhhF[i] = WhhT[(size_t)k0 * G_ + g];
  }
  for (int c = tid; c < NC; c += THREADS) {
    int g = (c / SLICE) * H_ + wg * SLICE + (c % SLICE);
    sBih[c] = bih[g];
    sBhh[c] = bhh[g];
  }
  __syncthreads();

  // Prologue: TDM-load x_0 and project it (off the h critical chain).
  if (wave == 0) {
    tdm_load_tile(x, SX_LDS_OFF);
    __builtin_amdgcn_s_wait_tensorcnt(0);
  }
  __syncthreads();
  proj3(sX, (const v16bf*)sWihF, sXP, wave, lane);

  for (int t = 0; t < T_; ++t) {
    const float* hread  = hbuf + (size_t)(t & 1) * B_ * H_;
    float*       hwrite = hbuf + (size_t)((t + 1) & 1) * B_ * H_;

    // ---- critical chain: h(t-1) -> h-projection -> gates -> h(t) ----------
    for (int i = tid; i < (B_ * H_) / 4; i += THREADS) {  // f32 -> bf16 pack
      float4 f = ((const float4*)hread)[i];
      v4bf o;
      o[0] = (__bf16)f.x; o[1] = (__bf16)f.y; o[2] = (__bf16)f.z; o[3] = (__bf16)f.w;
      ((v4bf*)sH)[i] = o;
    }
    __syncthreads();                       // sH ready (and sXP from prev iter)
    proj3(sH, (const v16bf*)sWhhF, sHP, wave, lane);
    __syncthreads();                       // sHP ready

    for (int i = tid; i < B_ * SLICE; i += THREADS) {
      int b = i / SLICE, j = i % SLICE;
      int jg = wg * SLICE + j;
      float xr = sXP[b * NC + j]             + sBih[j];
      float xz = sXP[b * NC + SLICE + j]     + sBih[SLICE + j];
      float xn = sXP[b * NC + 2 * SLICE + j] + sBih[2 * SLICE + j];
      float hr = sHP[b * NC + j]             + sBhh[j];
      float hz = sHP[b * NC + SLICE + j]     + sBhh[SLICE + j];
      float hn = sHP[b * NC + 2 * SLICE + j] + sBhh[2 * SLICE + j];
      float rg = 1.0f / (1.0f + __expf(-(xr + hr)));
      float zg = 1.0f / (1.0f + __expf(-(xz + hz)));
      float ta = xn + rg * hn;                      // tanh via single exp
      float e2 = __expf(2.0f * ta);
      float ng = (e2 - 1.0f) / (e2 + 1.0f);
      float hp = hread[b * H_ + jg];
      float hv = (1.0f - zg) * ng + zg * hp;
      hwrite[b * H_ + jg] = hv;
      if (yf)  yf[((size_t)b * T_ + t) * H_ + jg] = hv;
      if (ybf) ybf[((size_t)b * T_ + t) * H_ + jg] = (__bf16)hv;
    }
    __threadfence();
    __syncthreads();
    if (tid == 0)                          // arrive early ...
      __hip_atomic_fetch_add(cnt, 1u, __ATOMIC_RELEASE, __HIP_MEMORY_SCOPE_AGENT);

    // ---- barrier shadow: TDM-stage + project x(t+1) while others arrive ---
    if (t + 1 < T_) {
      if (wave == 0) {
        tdm_load_tile(x + (size_t)(t + 1) * D_, SX_LDS_OFF);
        __builtin_amdgcn_s_wait_tensorcnt(0);
      } else if (wave == 1 && t + 2 < T_) {
        for (int i = lane; i < (B_ * D_) / 8; i += 32)  // prefetch t+2 rows
          __builtin_prefetch(
              (const char*)(x + ((size_t)(i >> 5) * T_ + (t + 2)) * D_) +
                  (i & 31) * 16, 0, 1);
      }
      __syncthreads();
      proj3(sX, (const v16bf*)sWihF, sXP, wave, lane);
    }

    // ---- complete the cross-WG step barrier ------------------------------
    if (wave == 0)
      __builtin_amdgcn_s_cluster_barrier();  // s_barrier_signal/-wait -3; NOP
                                             // when not cluster-dispatched
    if (tid == 0) {
      const unsigned target = (unsigned)NWG * (unsigned)(t + 1);
      while (__hip_atomic_load(cnt, __ATOMIC_ACQUIRE, __HIP_MEMORY_SCOPE_AGENT) < target) {
        __builtin_amdgcn_s_sleep(1);
      }
    }
    __syncthreads();
    __threadfence();
  }
}

// ----------------------------------------------------------------- launch ---
extern "C" void kernel_launch(void* const* d_in, const int* in_sizes, int n_in,
                              void* d_out, int out_size, void* d_ws, size_t ws_size,
                              hipStream_t stream) {
  const float* x   = (const float*)d_in[0];
  const float* Wih = (const float*)d_in[1];
  const float* Whh = (const float*)d_in[2];
  const float* bih = (const float*)d_in[3];
  const float* bhh = (const float*)d_in[4];
  float* out = (float*)d_out;

  char* ws = (char*)d_ws;
  size_t off = 0;
  __bf16* WihT = (__bf16*)(ws + off); off += (size_t)L_ * D_ * G_ * sizeof(__bf16);
  __bf16* WhhT = (__bf16*)(ws + off); off += (size_t)L_ * H_ * G_ * sizeof(__bf16);
  float*  hbuf = (float*)(ws + off);  off += (size_t)2 * B_ * H_ * sizeof(float);
  unsigned* cnt = (unsigned*)(ws + off); off += 256;
  __bf16* xbf  = (__bf16*)(ws + off); off += (size_t)B_ * T_ * D_ * sizeof(__bf16);
  __bf16* ybf  = (__bf16*)(ws + off); off += (size_t)B_ * T_ * H_ * sizeof(__bf16);

  {
    const int total = L_ * G_ * D_;
    gru_wcvt<<<(total + 255) / 256, 256, 0, stream>>>(Wih, Whh, WihT, WhhT);
  }
  {
    const int n4 = (B_ * T_ * D_) / 4;
    gru_xcvt<<<(n4 + 255) / 256, 256, 0, stream>>>(x, xbf, n4);
  }

  const size_t lds_bytes =
      (size_t)(2 * NT * KK * 32 * 16 + B_ * D_ + B_ * H_) * sizeof(__bf16) +
      (size_t)(2 * B_ * NC + 2 * NC) * sizeof(float);

  for (int l = 0; l < L_; ++l) {
    gru_init<<<(2 * B_ * H_ + 255) / 256, 256, 0, stream>>>(hbuf, cnt);
    const __bf16* xin = (l == 0) ? xbf : ybf;
    float*  yf_out  = (l == L_ - 1) ? out : nullptr;
    __bf16* ybf_out = (l == L_ - 1) ? nullptr : ybf;
    gru_scan<<<NWG, THREADS, lds_bytes, stream>>>(
        xin, WihT + (size_t)l * D_ * G_, WhhT + (size_t)l * H_ * G_,
        bih + (size_t)l * G_, bhh + (size_t)l * G_, hbuf, cnt, yf_out, ybf_out);
  }
}
